// GroupedQueryAttentionBranch_1623497638713
// MI455X (gfx1250) — matrix operand order repaired and checked by reading, not compiled, not run on hardware
//
#include <hip/hip_runtime.h>
#include <hip/hip_bf16.h>
#include <stdint.h>

// ---- CDNA5 WMMA types ----
typedef __attribute__((ext_vector_type(16))) __bf16         v16bf;
typedef __attribute__((ext_vector_type(8)))  float          v8f;
typedef __attribute__((ext_vector_type(4)))  unsigned int   u32x4;
typedef __attribute__((ext_vector_type(4)))  float          f32x4;
typedef __attribute__((ext_vector_type(4)))  unsigned short u16x4;

#define PTOK 2048
#define EDIM 768
#define HDIM 64
#define NQ   12
#define NALL (NQ*HDIM + 2*HDIM)   // 896 fused output features (q heads | k | v)
#define NKSTEP (EDIM/32)          // 24 k-steps for E=768 GEMMs
#define LDSS 2052                 // padded scores row stride in floats (4-bank skew)
#define ALDP 776                  // padded A-stage row stride in ushorts (1552B -> 4-bank skew)

union FragB { v16bf v; unsigned short u[16]; u32x4 q[2]; };

__device__ __forceinline__ unsigned short f2bf(float f) {
  uint32_t u = __float_as_uint(f);
  u += 0x7FFFu + ((u >> 16) & 1u);      // round-to-nearest-even
  return (unsigned short)(u >> 16);
}

// A-fragment (16x32, MxK) from row-major bf16 in global memory (ISA 7.12.2):
// lanes 0-15: M=lane, VGPR0-3 K=h*8+0..7 ; VGPR4-7 K=16+h*8+0..7 (h = lane>>4)
__device__ __forceinline__ v16bf load_a_frag(const unsigned short* base, int ld, int m0, int k0) {
  int lane = threadIdx.x & 31;
  const unsigned short* p = base + (size_t)(m0 + (lane & 15)) * ld + k0 + (lane >> 4) * 8;
  FragB f;
  f.q[0] = *(const u32x4*)(p);
  f.q[1] = *(const u32x4*)(p + 16);
  return f.v;
}

// Same A-fragment but sourced from an LDS-staged row-block (padded stride).
__device__ __forceinline__ v16bf load_a_frag_lds(const unsigned short* sA, int k0) {
  int lane = threadIdx.x & 31;
  const unsigned short* p = sA + (lane & 15) * ALDP + k0 + (lane >> 4) * 8;
  FragB f;
  f.q[0] = *(const u32x4*)(p);
  f.q[1] = *(const u32x4*)(p + 16);
  return f.v;
}

// B-fragment (32x16, KxN) loaded from B-transposed (row-major [N,K]):
// lane n holds column N=n0+(lane&15); VGPR0-3 K=h*8+0..7 ; VGPR4-7 K=16+h*8+0..7 (h = lane>>4)
__device__ __forceinline__ v16bf load_b_frag(const unsigned short* bt, int ld, int n0, int k0) {
  int lane = threadIdx.x & 31;
  const unsigned short* p = bt + (size_t)(n0 + (lane & 15)) * ld + k0 + (lane >> 4) * 8;
  FragB f;
  f.q[0] = *(const u32x4*)(p);
  f.q[1] = *(const u32x4*)(p + 16);
  return f.v;
}

__device__ __forceinline__ v8f wmma_bf16(v16bf a, v16bf b, v8f c) {
  return __builtin_amdgcn_wmma_f32_16x16x32_bf16(false, a, false, b, (short)0, c, false, false);
}

// Cooperative stage of a 16 x EDIM bf16 row-block into padded LDS.
__device__ __forceinline__ void stage_rowblock(unsigned short* sA,
                                               const unsigned short* __restrict__ g,
                                               int m0) {
  for (int idx = threadIdx.x; idx < 16 * (EDIM / 8); idx += blockDim.x) {
    int r  = idx / (EDIM / 8);
    int cc = idx % (EDIM / 8);
    *(u32x4*)&sA[r * ALDP + cc * 8] =
        *(const u32x4*)(g + (size_t)(m0 + r) * EDIM + cc * 8);
  }
}

// ---- f32 -> bf16 conversion (vectorized: float4 in, ushort4 out) ----
__global__ void cvt_bf16_kernel(const float* __restrict__ src, unsigned short* __restrict__ dst, int n) {
  int i = (blockIdx.x * blockDim.x + threadIdx.x) * 4;
  if (i < n) {
    f32x4 v = *(const f32x4*)(src + i);
    u16x4 o;
#pragma unroll
    for (int j = 0; j < 4; ++j) o[j] = f2bf(v[j]);
    *(u16x4*)(dst + i) = o;
  }
}

// ---- fused QKV projection: [2048,768] x [768,896] with bias, scattered bf16 stores ----
// Loop nest: outer k-steps, inner 7 N-tiles per wave with independent accumulators,
// so each k-step issues 7 independent B loads feeding 7 independent WMMA chains.
// launch_bounds min-waves=1: allow the allocator to keep several B buffers live.
#define QKV_TPW 7   // tiles per wave: 56 tiles / 8 waves
__launch_bounds__(256, 1)
__global__ void qkv_proj_kernel(const unsigned short* __restrict__ xbf,
                                const unsigned short* __restrict__ wall,
                                const float* __restrict__ bq, const float* __restrict__ bk,
                                const float* __restrict__ bv,
                                unsigned short* __restrict__ qbf,   // [NQ, P, H]
                                unsigned short* __restrict__ kbf,   // [P, H]
                                unsigned short* __restrict__ vtbf)  // [H, P] (transposed)
{
  __shared__ unsigned short sA[16 * ALDP];   // x row-block, ~24.8 KB
  int m0   = blockIdx.x * 16;
  int wave = threadIdx.x >> 5;               // 0..7
  int lane = threadIdx.x & 31;
  int r8   = (lane >> 4) * 8;
  int nc   = lane & 15;

  stage_rowblock(sA, xbf, m0);
  __syncthreads();

  v8f acc[QKV_TPW];
#pragma unroll
  for (int j = 0; j < QKV_TPW; ++j) acc[j] = v8f{};

  for (int i = 0; i < NKSTEP; ++i) {
    v16bf a = load_a_frag_lds(sA, i * 32);
    // issue the whole independent B batch before consuming any of it
    v16bf bb[QKV_TPW];
#pragma unroll
    for (int j = 0; j < QKV_TPW; ++j)
      bb[j] = load_b_frag(wall, EDIM, (wave + 8 * j) * 16, i * 32);
#pragma unroll
    for (int j = 0; j < QKV_TPW; ++j)
      acc[j] = wmma_bf16(a, bb[j], acc[j]);
  }

#pragma unroll
  for (int j = 0; j < QKV_TPW; ++j) {
    int n0 = (wave + 8 * j) * 16;
    int n  = n0 + nc;
    float bias = (n < NQ*HDIM) ? bq[n]
               : (n < NQ*HDIM + HDIM) ? bk[n - NQ*HDIM]
               : bv[n - NQ*HDIM - HDIM];
#pragma unroll
    for (int r = 0; r < 8; ++r) {
      int row = m0 + r + r8;
      unsigned short val = f2bf(acc[j][r] + bias);
      if (n < NQ*HDIM) {
        qbf[((size_t)(n >> 6) * PTOK + row) * HDIM + (n & 63)] = val;
      } else if (n < NQ*HDIM + HDIM) {
        kbf[(size_t)row * HDIM + (n - NQ*HDIM)] = val;
      } else {
        vtbf[(size_t)(n - NQ*HDIM - HDIM) * PTOK + row] = val;  // transposed
      }
    }
  }
}

// ---- attention: scores -> softmax -> probs(f32 NT out) -> probs @ V -> merged(bf16) ----
// 128 KB LDS caps residency at 2 WGs/WGP (2 waves/SIMD), so per-wave memory-level
// parallelism is the only latency-hiding lever: min-waves=1 frees the VGPR budget.
__launch_bounds__(128, 1)
__global__ void attn_kernel(const unsigned short* __restrict__ qbf,
                            const unsigned short* __restrict__ kbf,
                            const unsigned short* __restrict__ vtbf,
                            float* __restrict__ probs,           // [NQ, P, P]
                            unsigned short* __restrict__ merged) // [P, NQ*H]
{
  __shared__ float s[16 * LDSS];   // exp-scores, row-padded (~128 KB)
  __shared__ float red[16 * 8];
  __shared__ float rmax[16];
  __shared__ float rsum[16];

  int m0   = blockIdx.x * 16;
  int head = blockIdx.y;
  int wave = threadIdx.x >> 5;
  int lane = threadIdx.x & 31;
  int half = lane >> 4;
  int nc   = lane & 15;

  // Phase 1: scores tile [16 x 2048] = Q[16x64] * K^T, scaled, into LDS.
  // Groups of 4 K-tiles: 8 independent B loads -> 4 independent WMMA chains.
  const unsigned short* qhead = qbf + (size_t)head * PTOK * HDIM;
  v16bf aq0 = load_a_frag(qhead, HDIM, m0, 0);
  v16bf aq1 = load_a_frag(qhead, HDIM, m0, 32);
  {
    int tbase = wave * 32;
    for (int tg = 0; tg < 32; tg += 4) {
      v16bf bb[8];
#pragma unroll
      for (int u = 0; u < 4; ++u) {
        bb[2*u]     = load_b_frag(kbf, HDIM, (tbase + tg + u) * 16, 0);
        bb[2*u + 1] = load_b_frag(kbf, HDIM, (tbase + tg + u) * 16, 32);
      }
      v8f cc[4];
#pragma unroll
      for (int u = 0; u < 4; ++u) {
        v8f c = {};
        c = wmma_bf16(aq0, bb[2*u], c);
        c = wmma_bf16(aq1, bb[2*u + 1], c);
        cc[u] = c;
      }
#pragma unroll
      for (int u = 0; u < 4; ++u) {
        int n0 = (tbase + tg + u) * 16;
#pragma unroll
        for (int r = 0; r < 8; ++r)
          s[(r + half * 8) * LDSS + n0 + nc] = cc[u][r] * 0.125f;  // 1/sqrt(64)
      }
    }
  }
  __syncthreads();

  // Phase 2: row softmax stats (8 threads per row, stride-8 interleave)
  int rr = threadIdx.x >> 3;
  int g  = threadIdx.x & 7;
  float lmax = -3.0e38f;
  for (int j = 0; j < PTOK / 8; ++j)
    lmax = fmaxf(lmax, s[rr * LDSS + g + 8 * j]);
  red[rr * 8 + g] = lmax;
  __syncthreads();
  if (g == 0) {
    float m = red[rr * 8];
#pragma unroll
    for (int i = 1; i < 8; ++i) m = fmaxf(m, red[rr * 8 + i]);
    rmax[rr] = m;
  }
  __syncthreads();
  float mrow = rmax[rr];
  float lsum = 0.f;
  for (int j = 0; j < PTOK / 8; ++j) {
    int idx = rr * LDSS + g + 8 * j;
    float e = __expf(s[idx] - mrow);
    s[idx] = e;
    lsum += e;
  }
  red[rr * 8 + g] = lsum;
  __syncthreads();
  if (g == 0) {
    float t = 0.f;
#pragma unroll
    for (int i = 0; i < 8; ++i) t += red[rr * 8 + i];
    rsum[rr] = 1.0f / t;
  }
  __syncthreads();

  // Phase 3: normalized probs -> global f32. Dominant data movement (201 MB):
  // 16-byte vector stores, non-temporal so the write-once stream doesn't evict
  // the hot K/V/Q working set (~11 MB) from the 192 MB L2.
  for (int idx = threadIdx.x * 4; idx < 16 * PTOK; idx += 128 * 4) {
    int r = idx >> 11;
    int c = idx & (PTOK - 1);
    f32x4 v = *(const f32x4*)&s[r * LDSS + c];
    float rs = rsum[r];
#pragma unroll
    for (int j = 0; j < 4; ++j) v[j] *= rs;
    __builtin_nontemporal_store(v,
        (f32x4*)(probs + ((size_t)head * PTOK + m0 + r) * PTOK + c));
  }

  // Phase 4: attn = probs @ V ; wave owns 16 head-dim columns; A built from LDS.
  // Two interleaved accumulation chains so each chain's loads/A-build overlap
  // the other chain's WMMA.
  float rinv = rsum[nc];
  int n0 = wave * 16;
  v8f c0 = {}, c1 = {};
  for (int ks = 0; ks < PTOK; ks += 64) {
    v16bf vb0 = load_b_frag(vtbf, PTOK, n0, ks);
    v16bf vb1 = load_b_frag(vtbf, PTOK, n0, ks + 32);
    FragB a0, a1;
    const float* sp0 = &s[nc * LDSS + ks + half * 8];
    const float* sp1 = sp0 + 32;
#pragma unroll
    for (int j = 0; j < 8; ++j) { a0.u[j] = f2bf(sp0[j] * rinv);      a1.u[j] = f2bf(sp1[j] * rinv); }
#pragma unroll
    for (int j = 0; j < 8; ++j) { a0.u[8+j] = f2bf(sp0[16+j] * rinv); a1.u[8+j] = f2bf(sp1[16+j] * rinv); }
    c0 = wmma_bf16(a0.v, vb0, c0);
    c1 = wmma_bf16(a1.v, vb1, c1);
  }
#pragma unroll
  for (int r = 0; r < 8; ++r) {
    int row = m0 + r + half * 8;
    merged[(size_t)row * EDIM + head * HDIM + n0 + nc] = f2bf(c0[r] + c1[r]);  // 'q p h -> p (q h)'
  }
}

// ---- output projection: [2048,768] x [768,64] + bo -> f32 out ----
// merged row-block staged in LDS; two interleaved accumulation chains.
__launch_bounds__(128, 1)
__global__ void out_proj_kernel(const unsigned short* __restrict__ merged,
                                const unsigned short* __restrict__ wobf,
                                const float* __restrict__ bo,
                                float* __restrict__ out) {
  __shared__ unsigned short sA[16 * ALDP];
  int m0   = blockIdx.x * 16;
  int wave = threadIdx.x >> 5;
  int lane = threadIdx.x & 31;
  int n0   = wave * 16;

  stage_rowblock(sA, merged, m0);
  __syncthreads();

  v8f c0 = {}, c1 = {};
#pragma unroll
  for (int i = 0; i < NKSTEP; i += 2) {
    v16bf a0 = load_a_frag_lds(sA, i * 32);
    v16bf b0 = load_b_frag(wobf, EDIM, n0, i * 32);
    v16bf a1 = load_a_frag_lds(sA, (i + 1) * 32);
    v16bf b1 = load_b_frag(wobf, EDIM, n0, (i + 1) * 32);
    c0 = wmma_bf16(a0, b0, c0);
    c1 = wmma_bf16(a1, b1, c1);
  }
  int n = n0 + (lane & 15);
  float bias = bo[n];
#pragma unroll
  for (int r = 0; r < 8; ++r) {
    int row = m0 + r + (lane >> 4) * 8;
    out[(size_t)row * HDIM + n] = c0[r] + c1[r] + bias;
  }
}

extern "C" void kernel_launch(void* const* d_in, const int* in_sizes, int n_in,
                              void* d_out, int out_size, void* d_ws, size_t ws_size,
                              hipStream_t stream) {
  const float* x  = (const float*)d_in[0];
  const float* Wq = (const float*)d_in[1];
  const float* bq = (const float*)d_in[2];
  const float* Wk = (const float*)d_in[3];
  const float* bk = (const float*)d_in[4];
  const float* Wv = (const float*)d_in[5];
  const float* bv = (const float*)d_in[6];
  const float* Wo = (const float*)d_in[7];
  const float* bo = (const float*)d_in[8];

  unsigned short* ws = (unsigned short*)d_ws;
  size_t off = 0;
  unsigned short* xbf    = ws + off; off += (size_t)PTOK * EDIM;       // x bf16
  unsigned short* wall   = ws + off; off += (size_t)NALL * EDIM;       // [Wq;Wk;Wv] bf16
  unsigned short* wobf   = ws + off; off += (size_t)HDIM * EDIM;       // Wo bf16
  unsigned short* qbf    = ws + off; off += (size_t)NQ * PTOK * HDIM;  // Q bf16
  unsigned short* kbf    = ws + off; off += (size_t)PTOK * HDIM;       // K bf16
  unsigned short* vtbf   = ws + off; off += (size_t)HDIM * PTOK;       // V^T bf16
  unsigned short* merged = ws + off; off += (size_t)PTOK * EDIM;       // attn merged bf16

  float* out_ptr = (float*)d_out;                    // out: [1, 2048, 64]
  float* probs   = out_ptr + (size_t)PTOK * HDIM;    // probs: [12, 2048, 2048]

  // bf16 conversions (all sizes are multiples of 4)
  {
    int n;
    n = PTOK * EDIM;      cvt_bf16_kernel<<<(n/4 + 255) / 256, 256, 0, stream>>>(x,  xbf,  n);
    n = NQ * HDIM * EDIM; cvt_bf16_kernel<<<(n/4 + 255) / 256, 256, 0, stream>>>(Wq, wall, n);
    n = HDIM * EDIM;
    cvt_bf16_kernel<<<(n/4 + 255) / 256, 256, 0, stream>>>(Wk, wall + (size_t)NQ * HDIM * EDIM, n);
    cvt_bf16_kernel<<<(n/4 + 255) / 256, 256, 0, stream>>>(Wv, wall + (size_t)(NQ * HDIM + HDIM) * EDIM, n);
    cvt_bf16_kernel<<<(n/4 + 255) / 256, 256, 0, stream>>>(Wo, wobf, n);
  }

  qkv_proj_kernel<<<PTOK / 16, 256, 0, stream>>>(xbf, wall, bq, bk, bv, qbf, kbf, vtbf);
  attn_kernel<<<dim3(PTOK / 16, NQ), 128, 0, stream>>>(qbf, kbf, vtbf, probs, merged);
  out_proj_kernel<<<PTOK / 16, 128, 0, stream>>>(merged, wobf, bo, out_ptr);
}
